// GAT_63161789055110
// MI455X (gfx1250) — matrix operand order repaired and checked by reading, not compiled, not run on hardware
//
#include <hip/hip_runtime.h>
#include <math.h>

typedef __attribute__((ext_vector_type(16))) _Float16 v16h;
typedef __attribute__((ext_vector_type(8)))  float    v8f;

#define NNODES 50000
#define NEDGES 1600000
#define NEG_SLOPE 0.2f
#define EPSV 1e-16f

// ---- order-preserving float<->uint encoding for atomicMax-based segment max ----
__device__ __forceinline__ unsigned fenc(float f) {
  unsigned u = __float_as_uint(f);
  return (u & 0x80000000u) ? ~u : (u | 0x80000000u);
}
__device__ __forceinline__ float fdec(unsigned e) {
  return (e & 0x80000000u) ? __uint_as_float(e & 0x7fffffffu)
                           : __uint_as_float(~e);
}

__device__ __forceinline__ float lrelu(float v) {
  return v > 0.0f ? v : NEG_SLOPE * v;
}

// ---------------- WMMA GEMM: C[N, NC] = A[N, K] @ W[K, NC] -------------------
// One wave computes a 16x16 tile; blockDim = 32*(NC/16); grid = N/16 (N % 16 == 0).
// W is staged once per block into LDS as f16; A is streamed with b128 loads.
template <int K, int NC>
__global__ void k_gemm(const float* __restrict__ A, const float* __restrict__ W,
                       float* __restrict__ C) {
  __shared__ _Float16 lw[K * NC];

  // cooperative stage of W (f32 -> f16), vectorized
  for (int idx = threadIdx.x; idx < (K * NC) / 4; idx += blockDim.x) {
    float4 w4 = ((const float4*)W)[idx];
    lw[4 * idx + 0] = (_Float16)w4.x;
    lw[4 * idx + 1] = (_Float16)w4.y;
    lw[4 * idx + 2] = (_Float16)w4.z;
    lw[4 * idx + 3] = (_Float16)w4.w;
  }
  __syncthreads();

  const int lane = threadIdx.x & 31;
  const int wv   = threadIdx.x >> 5;          // N-tile index
  const int r    = lane & 15;
  const int kg   = lane >> 4;                 // lane half
  const int row  = blockIdx.x * 16 + r;       // A row (M)
  const int col  = wv * 16 + r;               // B/C col (N)
  const float4* xr = (const float4*)(A + (size_t)row * K);

  v8f acc = {};
#pragma unroll
  for (int kc = 0; kc < K; kc += 32) {
    // A (16x32 f16): lane needs contiguous K runs [kc+kg*8, +8) and [kc+16+kg*8, +8)
    const int q0 = (kc + kg * 8) >> 2;        // float4 index of first run
    float4 a0 = xr[q0];
    float4 a1 = xr[q0 + 1];
    float4 a2 = xr[q0 + 4];                   // +16 floats
    float4 a3 = xr[q0 + 5];
    v16h a, b;
    a[0]  = (_Float16)a0.x; a[1]  = (_Float16)a0.y;
    a[2]  = (_Float16)a0.z; a[3]  = (_Float16)a0.w;
    a[4]  = (_Float16)a1.x; a[5]  = (_Float16)a1.y;
    a[6]  = (_Float16)a1.z; a[7]  = (_Float16)a1.w;
    a[8]  = (_Float16)a2.x; a[9]  = (_Float16)a2.y;
    a[10] = (_Float16)a2.z; a[11] = (_Float16)a2.w;
    a[12] = (_Float16)a3.x; a[13] = (_Float16)a3.y;
    a[14] = (_Float16)a3.z; a[15] = (_Float16)a3.w;
    // B (32x16 f16, column per lane): same K runs as A: [kc+kg*8, +8) and [kc+16+kg*8, +8)
#pragma unroll
    for (int v = 0; v < 8; ++v) {
      const int kb = kc + kg * 8 + 2 * v + (v >= 4 ? 8 : 0);
      b[2 * v]     = lw[kb * NC + col];
      b[2 * v + 1] = lw[(kb + 1) * NC + col];
    }
    acc = __builtin_amdgcn_wmma_f32_16x16x32_f16(false, a, false, b,
                                                 (short)0, acc, false, false);
  }
  // C/D layout: lane half selects M 0-7 vs 8-15, VGPR v = M offset, N = lane&15
  const int mb = blockIdx.x * 16 + kg * 8;
#pragma unroll
  for (int v = 0; v < 8; ++v) C[(mb + v) * NC + col] = acc[v];
}

// ---------------- layer-1 attention coefficients (8 heads, dim 8) ------------
__global__ void k_coef1(const float* __restrict__ h1, const float* __restrict__ asrc,
                        const float* __restrict__ adst, float* __restrict__ s,
                        float* __restrict__ d) {
  int i = blockIdx.x * blockDim.x + threadIdx.x;   // i = n*8 + h
  if (i >= NNODES * 8) return;
  const int h = i & 7;
  float ss = 0.f, dd = 0.f;
#pragma unroll
  for (int k = 0; k < 8; ++k) {
    float v = h1[8 * i + k];                       // h1[n*64 + h*8 + k]
    ss += v * asrc[h * 8 + k];
    dd += v * adst[h * 8 + k];
  }
  s[i] = ss;
  d[i] = dd;
}

__device__ __forceinline__ void edge_nodes(int e, const long long* __restrict__ ei,
                                           int& sn, int& dn) {
  if (e < NEDGES) {
    sn = (int)ei[e];
    dn = (int)ei[NEDGES + e];
  } else {
    sn = dn = e - NEDGES;                          // self loop
  }
}

// ---------------- layer-1 edge passes (thread per edge*head) -----------------
__global__ void k_emax1(const long long* __restrict__ ei, const float* __restrict__ s,
                        const float* __restrict__ d, unsigned* __restrict__ m, int ET) {
  int i = blockIdx.x * blockDim.x + threadIdx.x;
  if (i >= ET * 8) return;
  int sn, dn; edge_nodes(i >> 3, ei, sn, dn);
  const int h = i & 7;
  float v = lrelu(s[sn * 8 + h] + d[dn * 8 + h]);
  atomicMax(&m[dn * 8 + h], fenc(v));
}

__global__ void k_esum1(const long long* __restrict__ ei, const float* __restrict__ s,
                        const float* __restrict__ d, const unsigned* __restrict__ m,
                        float* __restrict__ den, int ET) {
  int i = blockIdx.x * blockDim.x + threadIdx.x;
  if (i >= ET * 8) return;
  int sn, dn; edge_nodes(i >> 3, ei, sn, dn);
  const int h = i & 7;
  float v = lrelu(s[sn * 8 + h] + d[dn * 8 + h]);
  atomicAdd(&den[dn * 8 + h], __expf(v - fdec(m[dn * 8 + h])));
}

__global__ void k_eagg1(const long long* __restrict__ ei, const float* __restrict__ s,
                        const float* __restrict__ d, const unsigned* __restrict__ m,
                        const float* __restrict__ den, const float* __restrict__ h1,
                        float* __restrict__ agg, int ET) {
  int i = blockIdx.x * blockDim.x + threadIdx.x;
  if (i >= ET * 8) return;
  int sn, dn; edge_nodes(i >> 3, ei, sn, dn);
  const int h = i & 7;
  float v  = lrelu(s[sn * 8 + h] + d[dn * 8 + h]);
  float ex = __expf(v - fdec(m[dn * 8 + h]));
  float alpha = ex / (den[dn * 8 + h] + EPSV);
#pragma unroll
  for (int k = 0; k < 8; ++k)
    atomicAdd(&agg[dn * 64 + h * 8 + k], h1[sn * 64 + h * 8 + k] * alpha);
}

// ---------------- bias + ELU -------------------------------------------------
__global__ void k_elubias(const float* __restrict__ agg, const float* __restrict__ b,
                          float* __restrict__ h2) {
  int i = blockIdx.x * blockDim.x + threadIdx.x;
  if (i >= NNODES * 64) return;
  float v = agg[i] + b[i & 63];
  h2[i] = v > 0.0f ? v : (__expf(v) - 1.0f);
}

// ---------------- layer-2 attention coefficients (1 head, dim 32) ------------
__global__ void k_coef2(const float* __restrict__ z, const float* __restrict__ asrc,
                        const float* __restrict__ adst, float* __restrict__ s,
                        float* __restrict__ d) {
  int n = blockIdx.x * blockDim.x + threadIdx.x;
  if (n >= NNODES) return;
  float ss = 0.f, dd = 0.f;
#pragma unroll
  for (int c = 0; c < 32; ++c) {
    float v = z[n * 32 + c];
    ss += v * asrc[c];
    dd += v * adst[c];
  }
  s[n] = ss;
  d[n] = dd;
}

// ---------------- layer-2 edge passes (thread per edge) ----------------------
__global__ void k_emax2(const long long* __restrict__ ei, const float* __restrict__ s,
                        const float* __restrict__ d, unsigned* __restrict__ m, int ET) {
  int e = blockIdx.x * blockDim.x + threadIdx.x;
  if (e >= ET) return;
  int sn, dn; edge_nodes(e, ei, sn, dn);
  atomicMax(&m[dn], fenc(lrelu(s[sn] + d[dn])));
}

__global__ void k_esum2(const long long* __restrict__ ei, const float* __restrict__ s,
                        const float* __restrict__ d, const unsigned* __restrict__ m,
                        float* __restrict__ den, int ET) {
  int e = blockIdx.x * blockDim.x + threadIdx.x;
  if (e >= ET) return;
  int sn, dn; edge_nodes(e, ei, sn, dn);
  float v = lrelu(s[sn] + d[dn]);
  atomicAdd(&den[dn], __expf(v - fdec(m[dn])));
}

__global__ void k_eagg2(const long long* __restrict__ ei, const float* __restrict__ s,
                        const float* __restrict__ d, const unsigned* __restrict__ m,
                        const float* __restrict__ den, const float* __restrict__ z,
                        float* __restrict__ agg, int ET) {
  int e = blockIdx.x * blockDim.x + threadIdx.x;
  if (e >= ET) return;
  int sn, dn; edge_nodes(e, ei, sn, dn);
  float v  = lrelu(s[sn] + d[dn]);
  float ex = __expf(v - fdec(m[dn]));
  float alpha = ex / (den[dn] + EPSV);
#pragma unroll
  for (int c = 0; c < 32; ++c)
    atomicAdd(&agg[dn * 32 + c], z[sn * 32 + c] * alpha);
}

// ---------------- final bias + log_softmax: one wave32 per node --------------
__global__ void k_lsm(const float* __restrict__ agg, const float* __restrict__ b,
                      float* __restrict__ out) {
  const int lane = threadIdx.x & 31;
  const int node = blockIdx.x * (blockDim.x >> 5) + (threadIdx.x >> 5);
  if (node >= NNODES) return;
  float v = agg[node * 32 + lane] + b[lane];
  float mx = v;
#pragma unroll
  for (int off = 16; off > 0; off >>= 1) mx = fmaxf(mx, __shfl_xor(mx, off, 32));
  float ex = __expf(v - mx);
  float sm = ex;
#pragma unroll
  for (int off = 16; off > 0; off >>= 1) sm += __shfl_xor(sm, off, 32);
  out[node * 32 + lane] = v - mx - logf(sm);
}

extern "C" void kernel_launch(void* const* d_in, const int* in_sizes, int n_in,
                              void* d_out, int out_size, void* d_ws, size_t ws_size,
                              hipStream_t stream) {
  const float*     x   = (const float*)d_in[0];
  const long long* ei  = (const long long*)d_in[1];
  const float*     W1  = (const float*)d_in[2];
  const float*     as1 = (const float*)d_in[3];
  const float*     ad1 = (const float*)d_in[4];
  const float*     b1  = (const float*)d_in[5];
  const float*     W2  = (const float*)d_in[6];
  const float*     as2 = (const float*)d_in[7];
  const float*     ad2 = (const float*)d_in[8];
  const float*     b2  = (const float*)d_in[9];
  float*           out = (float*)d_out;

  const int N = NNODES;
  const int ET = NEDGES + NNODES;   // edges + self loops

  // workspace layout (floats)
  float* ws = (float*)d_ws;
  float* h1  = ws; ws += (size_t)N * 64;
  float* s1  = ws; ws += (size_t)N * 8;
  float* dd1 = ws; ws += (size_t)N * 8;
  float* h2  = ws; ws += (size_t)N * 64;
  float* z2  = ws; ws += (size_t)N * 32;
  float* s2  = ws; ws += (size_t)N;
  float* dd2 = ws; ws += (size_t)N;
  float* zstart = ws;                       // ---- zero-initialized region ----
  unsigned* m1   = (unsigned*)ws; ws += (size_t)N * 8;
  float*    den1 = ws;            ws += (size_t)N * 8;
  float*    agg1 = ws;            ws += (size_t)N * 64;
  unsigned* m2   = (unsigned*)ws; ws += (size_t)N;
  float*    den2 = ws;            ws += (size_t)N;
  float*    agg2 = ws;            ws += (size_t)N * 32;
  size_t zbytes = (size_t)(ws - zstart) * sizeof(float);
  hipMemsetAsync(zstart, 0, zbytes, stream);   // graph-capturable

  const int T = 256;

  // ---- layer 1 ----
  k_gemm<128, 64><<<N / 16, 128, 0, stream>>>(x, W1, h1);
  k_coef1<<<(N * 8 + T - 1) / T, T, 0, stream>>>(h1, as1, ad1, s1, dd1);
  const int e8 = ET * 8;
  k_emax1<<<(e8 + T - 1) / T, T, 0, stream>>>(ei, s1, dd1, m1, ET);
  k_esum1<<<(e8 + T - 1) / T, T, 0, stream>>>(ei, s1, dd1, m1, den1, ET);
  k_eagg1<<<(e8 + T - 1) / T, T, 0, stream>>>(ei, s1, dd1, m1, den1, h1, agg1, ET);
  k_elubias<<<(N * 64 + T - 1) / T, T, 0, stream>>>(agg1, b1, h2);

  // ---- layer 2 ----
  k_gemm<64, 32><<<N / 16, 64, 0, stream>>>(h2, W2, z2);
  k_coef2<<<(N + T - 1) / T, T, 0, stream>>>(z2, as2, ad2, s2, dd2);
  k_emax2<<<(ET + T - 1) / T, T, 0, stream>>>(ei, s2, dd2, m2, ET);
  k_esum2<<<(ET + T - 1) / T, T, 0, stream>>>(ei, s2, dd2, m2, den2, ET);
  k_eagg2<<<(ET + T - 1) / T, T, 0, stream>>>(ei, s2, dd2, m2, den2, z2, agg2, ET);

  // ---- log_softmax ----
  k_lsm<<<(N + 7) / 8, 256, 0, stream>>>(agg2, b2, out);
}
